// Net_50190987821199
// MI455X (gfx1250) — matrix operand order repaired and checked by reading, not compiled, not run on hardware
//
#include <hip/hip_runtime.h>
#include <math.h>

// ---------------------------------------------------------------------------
// S2VT captioner for MI455X (gfx1250, wave32, WMMA).
// All GEMMs use v_wmma_f32_16x16x32_f16 (f16 in, f32 accumulate).
// Hoisted off the recurrent chain: encoder input projection (21.4 GFLOP),
// decoder word projection, bias sums, argmax targets. Recurrent cells are
// fused gates-GEMM + LSTM activation kernels (16 blocks x 256 thr).
// ---------------------------------------------------------------------------

typedef _Float16 f16;
typedef _Float16 v8h  __attribute__((ext_vector_type(8)));
typedef _Float16 v16h __attribute__((ext_vector_type(16)));
typedef float    v8f  __attribute__((ext_vector_type(8)));

#define B_    32
#define TENC  80
#define FEAT_ 4096
#define H_    256
#define G4    1024    // 4*H
#define W2V_  256
#define TDEC  30
#define NSTEP 29
#define V_    32000

__device__ __forceinline__ float sigm(float x) { return 1.0f / (1.0f + __expf(-x)); }

// --- WMMA micro-GEMM over K (multiple of 32) (A row-major [M][K], W row-major [N][K]) ----
__device__ __forceinline__ v8f wmma_k(const f16* __restrict__ Arow,
                                      const f16* __restrict__ Wrow,
                                      int K, int hi, v8f acc) {
  for (int k0 = 0; k0 < K; k0 += 32) {
    union { v16h v; v8h h[2]; } a, b;
    a.h[0] = *(const v8h*)(Arow + k0 + hi * 8);        // K = hi*8 .. +7
    a.h[1] = *(const v8h*)(Arow + k0 + hi * 8 + 16);   // K = hi*8+16 .. +23
    b.h[0] = *(const v8h*)(Wrow + k0 + hi * 8);        // K = hi*8 .. +7
    b.h[1] = *(const v8h*)(Wrow + k0 + hi * 8 + 16);   // K = hi*8+16 .. +23
    acc = __builtin_amdgcn_wmma_f32_16x16x32_f16(false, a.v, false, b.v,
                                                 (short)0, acc, false, false);
  }
  return acc;
}

// --- generic C[M][N] = A[M][K] @ W[N][K]^T + bias[N], one wave per 16x16 tile
__global__ void wmma_gemm_nt(const f16* __restrict__ A, const f16* __restrict__ W,
                             float* __restrict__ C, const float* __restrict__ bias,
                             int Mt, int Nt, int K, int ldo) {
  int wave = (blockIdx.x * blockDim.x + threadIdx.x) >> 5;
  int lane = threadIdx.x & 31;
  if (wave >= Mt * Nt) return;
  int mt = wave % Mt, nt = wave / Mt;
  int row = lane & 15, hi = lane >> 4;
  const f16* Arow = A + (size_t)(mt * 16 + row) * K;
  const f16* Wrow = W + (size_t)(nt * 16 + row) * K;
  v8f acc = {};
  acc = wmma_k(Arow, Wrow, K, hi, acc);
  int n = nt * 16 + row;
  float bv = bias ? bias[n] : 0.0f;
#pragma unroll
  for (int v = 0; v < 8; ++v) {
    int m = mt * 16 + v + hi * 8;
    C[(size_t)m * ldo + n] = acc[v] + bv;
  }
}

// --- fused LSTM cell: gates = base + A1@W1^T (+A2@W2^T); then i,f,g,o update
// grid = 16 blocks (16 hidden idx each), block = 256 thr (8 waves = 2 Mtiles x 4 gates)
__global__ void lstm_step(const float* __restrict__ base, int base_vec,
                          const f16* __restrict__ A1, const f16* __restrict__ W1,
                          const f16* __restrict__ A2, const f16* __restrict__ W2,
                          float* __restrict__ c_state,
                          float* __restrict__ h_out, f16* __restrict__ h16_out,
                          float* __restrict__ h_copy) {
  __shared__ float gates[2][4][16][16];   // [mtile][gate][m][j]
  int tid = threadIdx.x;
  int wave = tid >> 5, lane = tid & 31;
  int g = wave & 3, mt = wave >> 2;
  int row = lane & 15, hi = lane >> 4;
  int n0 = blockIdx.x * 16;
  int ncol = g * H_ + n0 + row;           // gate column 0..1023

  v8f acc = {};
  acc = wmma_k(A1 + (size_t)(mt * 16 + row) * H_, W1 + (size_t)ncol * H_, H_, hi, acc);
  if (A2)
    acc = wmma_k(A2 + (size_t)(mt * 16 + row) * H_, W2 + (size_t)ncol * H_, H_, hi, acc);

#pragma unroll
  for (int v = 0; v < 8; ++v) {
    int mloc = v + hi * 8;
    float bv = base_vec ? base[ncol]
                        : base[(size_t)(mt * 16 + mloc) * G4 + ncol];
    gates[mt][g][mloc][row] = acc[v] + bv;
  }
  __syncthreads();

  for (int p = tid; p < 512; p += 256) {  // 32 batch x 16 hidden
    int b = p >> 4, j = p & 15;
    int mt2 = b >> 4, ml = b & 15;
    float gi = gates[mt2][0][ml][j];
    float gf = gates[mt2][1][ml][j];
    float gg = gates[mt2][2][ml][j];
    float go = gates[mt2][3][ml][j];
    int n = n0 + j;
    float c_old = c_state[b * H_ + n];
    float c_new = sigm(gf) * c_old + sigm(gi) * tanhf(gg);
    float h_new = sigm(go) * tanhf(c_new);
    c_state[b * H_ + n] = c_new;
    h_out[b * H_ + n] = h_new;
    h16_out[b * H_ + n] = (f16)h_new;
    if (h_copy) h_copy[b * H_ + n] = h_new;
  }
}

// --- prologue conversions -------------------------------------------------
__global__ void cvt_f16(const float* __restrict__ s, f16* __restrict__ d, int n) {
  int i = blockIdx.x * 256 + threadIdx.x;
  if (i < n) d[i] = (f16)s[i];
}
// feat [B][80][4096] -> feat16 [t*32+b][4096]
__global__ void feat_perm(const float* __restrict__ s, f16* __restrict__ d) {
  int i = blockIdx.x * 256 + threadIdx.x;          // 2560*4096
  int k = i & 4095, r = i >> 12, t = r >> 5, b = r & 31;
  d[i] = (f16)s[((size_t)b * TENC + t) * FEAT_ + k];
}
// caption [B][30][256] -> cap16 [t*32+b][256], t<29
__global__ void cap_perm(const float* __restrict__ s, f16* __restrict__ d) {
  int i = blockIdx.x * 256 + threadIdx.x;          // 928*256
  int k = i & 255, r = i >> 8, t = r >> 5, b = r & 31;
  d[i] = (f16)s[((size_t)b * TDEC + t) * W2V_ + k];
}
// w_ih2 [1024][512] -> word half [1024][256] + h1 half [1024][256]
__global__ void split_ih2(const float* __restrict__ s, f16* __restrict__ dw,
                          f16* __restrict__ dh) {
  int i = blockIdx.x * 256 + threadIdx.x;          // 1024*256
  int r = i >> 8, k = i & 255;
  dw[i] = (f16)s[(size_t)r * 512 + k];
  dh[i] = (f16)s[(size_t)r * 512 + 256 + k];
}
__global__ void bias_sum(const float* a, const float* b, const float* c,
                         const float* d, float* o1, float* o2) {
  int i = blockIdx.x * 256 + threadIdx.x;
  if (i < G4) { o1[i] = a[i] + b[i]; o2[i] = c[i] + d[i]; }
}
// targets[t*32+b] = argmax_v onehot[b][t+1][v], one wave per row
__global__ void argmax_k(const float* __restrict__ onehot, int* __restrict__ tg) {
  int wave = (blockIdx.x * blockDim.x + threadIdx.x) >> 5;
  int lane = threadIdx.x & 31;
  if (wave >= NSTEP * B_) return;
  int t = wave >> 5, b = wave & 31;
  const float* row = onehot + ((size_t)b * TDEC + (t + 1)) * V_;
  float bv = -3.4e38f; int bi = 0;
  for (int v = lane; v < V_; v += 32) {
    float x = row[v];
    if (x > bv) { bv = x; bi = v; }
  }
  for (int off = 16; off; off >>= 1) {
    float ov = __shfl_xor(bv, off, 32);
    int   oi = __shfl_xor(bi, off, 32);
    if (ov > bv || (ov == bv && oi < bi)) { bv = ov; bi = oi; }
  }
  if (lane == 0) tg[t * B_ + b] = bi;
}

// --- CE: one block per batch row, log_softmax + gather ---------------------
__global__ void ce_row(const float* __restrict__ logits, const int* __restrict__ tgt,
                       float* __restrict__ ce_out) {
  int b = blockIdx.x;
  const float* row = logits + (size_t)b * V_;
  __shared__ float red[32];
  __shared__ float s_max, s_sum;
  int tid = threadIdx.x, lane = tid & 31, wave = tid >> 5;
  float mx = -3.4e38f;
  for (int v = tid; v < V_; v += 1024) mx = fmaxf(mx, row[v]);
  for (int off = 16; off; off >>= 1) mx = fmaxf(mx, __shfl_xor(mx, off, 32));
  if (lane == 0) red[wave] = mx;
  __syncthreads();
  if (tid < 32) {
    float m = red[tid];
    for (int off = 16; off; off >>= 1) m = fmaxf(m, __shfl_xor(m, off, 32));
    if (tid == 0) s_max = m;
  }
  __syncthreads();
  float sum = 0.0f;
  for (int v = tid; v < V_; v += 1024) sum += __expf(row[v] - s_max);
  for (int off = 16; off; off >>= 1) sum += __shfl_xor(sum, off, 32);
  if (lane == 0) red[wave] = sum;
  __syncthreads();
  if (tid < 32) {
    float s = red[tid];
    for (int off = 16; off; off >>= 1) s += __shfl_xor(s, off, 32);
    if (tid == 0) s_sum = s;
  }
  __syncthreads();
  if (tid == 0) ce_out[b] = (s_max + __logf(s_sum)) - row[tgt[b]];
}
__global__ void loss_acc(const float* __restrict__ ce, float* __restrict__ loss) {
  if (threadIdx.x == 0) {
    float a = 0.0f;
    for (int b = 0; b < B_; ++b) a += ce[b];      // ordered -> deterministic
    loss[0] += a * (1.0f / (B_ * B_));            // mean(ce)/B
  }
}

// --- attention: scores->softmax->context, replaces h2 ----------------------
__global__ void attn_kernel(const float* __restrict__ enc_h2,   // [80][32][256]
                            const float* __restrict__ h2n,      // [32][256]
                            float* __restrict__ h2_out, f16* __restrict__ h16_out) {
  __shared__ float sc[B_][TENC];
  int tid = threadIdx.x;
  for (int p = tid; p < B_ * TENC; p += 256) {
    int b = p / TENC, s = p % TENC;
    const float* e = enc_h2 + ((size_t)s * B_ + b) * H_;
    const float* h = h2n + (size_t)b * H_;
    float d = 0.0f;
    for (int k = 0; k < H_; ++k) d += e[k] * h[k];
    sc[b][s] = d;
  }
  __syncthreads();
  if (tid < B_) {
    float mx = -3.4e38f;
    for (int s = 0; s < TENC; ++s) mx = fmaxf(mx, sc[tid][s]);
    float sum = 0.0f;
    for (int s = 0; s < TENC; ++s) { float e = __expf(sc[tid][s] - mx); sc[tid][s] = e; sum += e; }
    float inv = 1.0f / sum;
    for (int s = 0; s < TENC; ++s) sc[tid][s] *= inv;
  }
  __syncthreads();
  for (int o = tid; o < B_ * H_; o += 256) {
    int b = o >> 8, k = o & 255;
    float a = 0.0f;
    for (int s = 0; s < TENC; ++s) a += sc[b][s] * enc_h2[((size_t)s * B_ + b) * H_ + k];
    h2_out[o] = a;
    h16_out[o] = (f16)a;
  }
}

__global__ void copy_loss(const float* __restrict__ loss, float* __restrict__ out) {
  if (threadIdx.x == 0 && blockIdx.x == 0) out[0] = loss[0];
}

// ---------------------------------------------------------------------------
extern "C" void kernel_launch(void* const* d_in, const int* in_sizes, int n_in,
                              void* d_out, int out_size, void* d_ws, size_t ws_size,
                              hipStream_t stream) {
  (void)in_sizes; (void)n_in; (void)out_size; (void)ws_size;
  const float* feat   = (const float*)d_in[0];
  const float* capt   = (const float*)d_in[1];
  const float* onehot = (const float*)d_in[2];
  // d_in[3] = keep_rate (unused, ==1)
  const float* w_ih1 = (const float*)d_in[4];
  const float* w_hh1 = (const float*)d_in[5];
  const float* b_ih1 = (const float*)d_in[6];
  const float* b_hh1 = (const float*)d_in[7];
  const float* w_ih2 = (const float*)d_in[8];
  const float* w_hh2 = (const float*)d_in[9];
  const float* b_ih2 = (const float*)d_in[10];
  const float* b_hh2 = (const float*)d_in[11];
  const float* w_out = (const float*)d_in[12];
  const float* b_out = (const float*)d_in[13];

  char* basep = (char*)d_ws;
  size_t off = 0;
  auto carve = [&](size_t bytes) -> char* {
    char* r = basep + off; off += (bytes + 255) & ~(size_t)255; return r;
  };
  f16*   feat16  = (f16*)carve((size_t)2560 * FEAT_ * 2);
  f16*   cap16   = (f16*)carve((size_t)928 * W2V_ * 2);
  f16*   wih1_16 = (f16*)carve((size_t)G4 * FEAT_ * 2);
  f16*   whh1_16 = (f16*)carve((size_t)G4 * H_ * 2);
  f16*   wword16 = (f16*)carve((size_t)G4 * W2V_ * 2);
  f16*   wh1_16  = (f16*)carve((size_t)G4 * H_ * 2);
  f16*   whh2_16 = (f16*)carve((size_t)G4 * H_ * 2);
  f16*   wout16  = (f16*)carve((size_t)V_ * H_ * 2);
  float* pre1    = (float*)carve((size_t)2560 * G4 * 4);
  float* preW    = (float*)carve((size_t)928 * G4 * 4);
  float* ench2   = (float*)carve((size_t)TENC * B_ * H_ * 4);
  float* logits  = (float*)carve((size_t)B_ * V_ * 4);
  float* bias1   = (float*)carve(G4 * 4);
  float* bias2   = (float*)carve(G4 * 4);
  int*   tgts    = (int*)carve(NSTEP * B_ * 4);
  // zeroed state region
  char* zstart = basep + off;
  float* c1      = (float*)carve(B_ * H_ * 4);
  float* c2      = (float*)carve(B_ * H_ * 4);
  f16*   h1_16_0 = (f16*)carve(B_ * H_ * 2);
  f16*   h1_16_1 = (f16*)carve(B_ * H_ * 2);
  f16*   h2_16_0 = (f16*)carve(B_ * H_ * 2);   // recurrent h2 / ctx
  f16*   h2_16_1 = (f16*)carve(B_ * H_ * 2);   // h2n scratch
  float* h1_32   = (float*)carve(B_ * H_ * 4); // f32 scratch (unread)
  float* h2s_32  = (float*)carve(B_ * H_ * 4); // f32 scratch / ctx
  float* h2n_32  = (float*)carve(B_ * H_ * 4); // decoder h2n (attention input)
  float* cebuf   = (float*)carve(B_ * 4);
  float* lossws  = (float*)carve(256);
  size_t zbytes = (size_t)((basep + off) - zstart);
  hipMemsetAsync(zstart, 0, zbytes, stream);

  f16* h1_16[2] = { h1_16_0, h1_16_1 };
  f16* h2_16[2] = { h2_16_0, h2_16_1 };

  // ---- prologue: f16 conversions, bias sums, targets ----
  feat_perm<<<40960, 256, 0, stream>>>(feat, feat16);
  cap_perm<<<928, 256, 0, stream>>>(capt, cap16);
  cvt_f16<<<16384, 256, 0, stream>>>(w_ih1, wih1_16, G4 * FEAT_);
  cvt_f16<<<1024, 256, 0, stream>>>(w_hh1, whh1_16, G4 * H_);
  cvt_f16<<<1024, 256, 0, stream>>>(w_hh2, whh2_16, G4 * H_);
  cvt_f16<<<32000, 256, 0, stream>>>(w_out, wout16, V_ * H_);
  split_ih2<<<1024, 256, 0, stream>>>(w_ih2, wword16, wh1_16);
  bias_sum<<<4, 256, 0, stream>>>(b_ih1, b_hh1, b_ih2, b_hh2, bias1, bias2);
  argmax_k<<<116, 256, 0, stream>>>(onehot, tgts);

  // ---- hoisted GEMMs (the bulk of the FLOPs) ----
  // pre1[2560][1024] = feat16 @ w_ih1^T + (b_ih1+b_hh1)
  wmma_gemm_nt<<<1280, 256, 0, stream>>>(feat16, wih1_16, pre1, bias1,
                                         160, 64, FEAT_, G4);
  // preW[928][1024] = cap16 @ w_ih2[:, :256]^T + (b_ih2+b_hh2)
  wmma_gemm_nt<<<464, 256, 0, stream>>>(cap16, wword16, preW, bias2,
                                        58, 64, W2V_, G4);

  // ---- encoder: 80 fused LSTM steps ----
  for (int t = 0; t < TENC; ++t) {
    int pi = t & 1, po = 1 - pi;
    lstm_step<<<16, 256, 0, stream>>>(pre1 + (size_t)t * B_ * G4, 0,
        h1_16[pi], whh1_16, nullptr, nullptr,
        c1, h1_32, h1_16[po], nullptr);
    lstm_step<<<16, 256, 0, stream>>>(bias2, 1,
        h2_16[pi], whh2_16, h1_16[po], wh1_16,
        c2, h2s_32, h2_16[po], ench2 + (size_t)t * B_ * H_);
  }
  // final h1 in h1_16[0], final h2 in h2_16[0]

  // ---- decoder: 29 steps (LSTM1, LSTM2, vocab GEMM, CE, attention) ----
  for (int d = 0; d < NSTEP; ++d) {
    int pi = d & 1, po = 1 - pi;
    lstm_step<<<16, 256, 0, stream>>>(bias1, 1,
        h1_16[pi], whh1_16, nullptr, nullptr,
        c1, h1_32, h1_16[po], nullptr);
    lstm_step<<<16, 256, 0, stream>>>(preW + (size_t)d * B_ * G4, 0,
        h2_16[0], whh2_16, h1_16[po], wh1_16,
        c2, h2n_32, h2_16[1], nullptr);
    // logits[32][32000] = h2n @ w_out^T + b_out
    wmma_gemm_nt<<<500, 256, 0, stream>>>(h2_16[1], wout16, logits, b_out,
                                          2, 2000, H_, V_);
    ce_row<<<32, 1024, 0, stream>>>(logits, tgts + d * B_, cebuf);
    loss_acc<<<1, 32, 0, stream>>>(cebuf, lossws);
    // ctx replaces h2 (written back to h2_16[0] for the next step)
    attn_kernel<<<1, 256, 0, stream>>>(ench2, h2n_32, h2s_32, h2_16[0]);
  }

  copy_loss<<<1, 32, 0, stream>>>(lossws, (float*)d_out);
}
